// _Memory_Block_74534862454886
// MI455X (gfx1250) — matrix operand order repaired and checked by reading, not compiled, not run on hardware
//
#include <hip/hip_runtime.h>
#include <hip/hip_bf16.h>
#include <math.h>

// ---------------- problem constants (from reference) ----------------
constexpr int NB = 8;      // batch
constexpr int CS = 512;    // channel dim (C*S)
constexpr int HW = 4096;   // tokens per sample (H*W)
constexpr int NA = 16;     // class dim
constexpr int NK = 256;    // nodes per class
constexpr float MAR = 0.999f;

typedef __attribute__((ext_vector_type(16))) __bf16 v16bf;
typedef __attribute__((ext_vector_type(8)))  float  v8f;
typedef __attribute__((ext_vector_type(4)))  unsigned int v4u;
typedef __attribute__((ext_vector_type(8)))  int    v8i_;
typedef __attribute__((ext_vector_type(4)))  int    v4i_;

#if defined(__has_builtin)
# if __has_builtin(__builtin_amdgcn_tensor_load_to_lds)
#  define HAVE_TDM 1
# endif
#endif

// LDS layout offsets (bytes) for the WMMA kernels
constexpr unsigned OFF_BFULL = 0;                       // [NK][CS] or [CS][NK] bf16 (256 KB)
constexpr unsigned OFF_ADEP  = NK * CS * 2;             // [CS][16] bf16 TDM deposit (16 KB)
constexpr unsigned OFF_SC    = OFF_ADEP;                // [16][NK] f32 score (overlays ADEP)
constexpr unsigned OFF_AS    = OFF_ADEP + 16384;        // [16][CS] bf16 token-major A (16 KB)
constexpr unsigned OFF_SF    = OFF_AS + 16384;          // [16][NK] bf16 softmax (8 KB, attn)
constexpr size_t   LDS_SCORE = OFF_AS + 16384;          // 294912
constexpr size_t   LDS_ATTN  = OFF_SF + 8192;           // 303104

__device__ __forceinline__ __bf16 f2bf(float f) { return (__bf16)f; }

// A-matrix (16x32 bf16): lane holds row M=lane&15, K = (lane/16)*16 + e
__device__ __forceinline__ int a_kloc(int e, int lane) {
    return ((lane >> 4) << 4) + e;                           // 0..31
}
// B-matrix (32x16 bf16): lane holds column N=lane&15, K = (lane/16)*16 + e
__device__ __forceinline__ int b_kloc(int e, int lane) {
    return ((lane >> 4) << 4) + e;                           // 0..31
}

#ifdef HAVE_TDM
// Issue one TDM 2D tile load: global -> LDS. D# bitfields per CDNA5 ISA 8.3/8.4.
__device__ __forceinline__ void tdm_load_2d(unsigned int lds_off, const void* gptr,
                                            unsigned int tile_d0, unsigned int tile_d1,
                                            unsigned int tens_d0, unsigned int tens_d1,
                                            unsigned long long stride0_elems,
                                            unsigned int data_size_code) {
    unsigned long long ga = (unsigned long long)gptr;
    v4u g0;
    g0[0] = 1u;                                   // count=1 (valid), user mode
    g0[1] = lds_off;                              // lds_addr (bytes)
    g0[2] = (unsigned int)(ga & 0xFFFFFFFFu);     // global_addr lo
    g0[3] = (unsigned int)((ga >> 32) & 0x01FFFFFFu) | (2u << 30);  // addr hi | type=2
    v8i_ g1;
    g1[0] = (int)((data_size_code & 3u) << 16);   // wg_mask=0, data_size
    g1[1] = (int)((tens_d0 & 0xFFFFu) << 16);     // tensor_dim0[15:0] @ bits63:48
    g1[2] = (int)(((tens_d0 >> 16) & 0xFFFFu) | ((tens_d1 & 0xFFFFu) << 16));
    g1[3] = (int)(((tens_d1 >> 16) & 0xFFFFu) | ((tile_d0 & 0xFFFFu) << 16));
    g1[4] = (int)(tile_d1 & 0xFFFFu);             // tile_dim1; tile_dim2=0
    g1[5] = (int)(stride0_elems & 0xFFFFFFFFull); // tensor_dim0_stride lo32
    g1[6] = (int)((stride0_elems >> 32) & 0xFFFFull);  // stride hi; dim1_stride=0
    g1[7] = 0;
    v4i_ gz = {0, 0, 0, 0};
#if __clang_major__ >= 23
    v8i_ gz8 = {0, 0, 0, 0, 0, 0, 0, 0};
    __builtin_amdgcn_tensor_load_to_lds(g0, g1, gz, gz, gz8, 0);
#else
    __builtin_amdgcn_tensor_load_to_lds(g0, g1, gz, gz, 0);
#endif
}
#endif

// ---------------- small kernels ----------------

__global__ void k_copy(const float* __restrict__ src, float* __restrict__ dst, int n) {
    int i = blockIdx.x * 256 + threadIdx.x;
    if (i < n) dst[i] = src[i];
}

// L2-normalize tokens of x along CS; store channel-major bf16: xnT[b][c][n]
__global__ void k_norm_x(const float* __restrict__ x, __bf16* __restrict__ xnT) {
    const int bpb = HW / 256;
    const int b  = blockIdx.x / bpb;
    const int n  = (blockIdx.x % bpb) * 256 + threadIdx.x;
    const float* xb = x + (size_t)b * CS * HW;
    float ss = 0.f;
    for (int c = 0; c < CS; ++c) {
        if (c + 8 < CS) __builtin_prefetch(&xb[(size_t)(c + 8) * HW + n], 0, 0);
        float v = xb[(size_t)c * HW + n];
        ss += v * v;
    }
    float inv = 1.0f / fmaxf(sqrtf(ss), 1e-12f);
    __bf16* o = xnT + (size_t)b * CS * HW;
    for (int c = 0; c < CS; ++c)
        o[(size_t)c * HW + n] = f2bf(xb[(size_t)c * HW + n] * inv);
}

// normalize rows of m[label[b]] -> mn bf16 [NK][CS]
__global__ void k_norm_m(const float* __restrict__ m, const int* __restrict__ label,
                         int b, __bf16* __restrict__ mn) {
    const int lab = label[b];
    const float* row = m + ((size_t)lab * NK + blockIdx.x) * CS;
    float v0 = row[threadIdx.x], v1 = row[threadIdx.x + 256];
    __shared__ float red[256];
    red[threadIdx.x] = v0 * v0 + v1 * v1;
    __syncthreads();
    for (int s = 128; s > 0; s >>= 1) {
        if (threadIdx.x < s) red[threadIdx.x] += red[threadIdx.x + s];
        __syncthreads();
    }
    float inv = 1.0f / fmaxf(sqrtf(red[0]), 1e-12f);
    __bf16* o = mn + (size_t)blockIdx.x * CS;
    o[threadIdx.x]       = f2bf(v0 * inv);
    o[threadIdx.x + 256] = f2bf(v1 * inv);
}

// ---------------- panel loaders (device helpers) ----------------

// Load bf16 panel [NK][CS] into OFF_BFULL, plus A tile [16 x CS] from
// channel-major xnT into token-major As[16][CS]. Ends fully synchronized.
__device__ __forceinline__ void load_panels(char* smem, const __bf16* panel,
                                            const __bf16* xb_n0, int t, int wv) {
    __bf16* Bfull = (__bf16*)(smem + OFF_BFULL);
    __bf16* Adep  = (__bf16*)(smem + OFF_ADEP);
    __bf16* As    = (__bf16*)(smem + OFF_AS);
#ifdef HAVE_TDM
    if (wv == 0) {
        tdm_load_2d((unsigned int)(size_t)Bfull, panel, CS, NK, CS, NK, CS, 1u);
        tdm_load_2d((unsigned int)(size_t)Adep, xb_n0, 16, CS, HW, CS, HW, 1u);
        __builtin_amdgcn_s_wait_tensorcnt(0);
    }
    __syncthreads();
    { // transpose [CS][16] deposit -> token-major As[16][CS]
        int c = t;
        #pragma unroll
        for (int m = 0; m < 16; ++m) As[m * CS + c] = Adep[c * 16 + m];
    }
    __syncthreads();
#else
    for (int i = t * 16; i < NK * CS; i += 512 * 16) {
        #pragma unroll
        for (int j = 0; j < 16; ++j) Bfull[i + j] = panel[i + j];
    }
    { int c = t;
      #pragma unroll
      for (int m = 0; m < 16; ++m) As[m * CS + c] = xb_n0[(size_t)c * HW + m];
    }
    __syncthreads();
#endif
}

// Reload bf16 panel [d1 rows][d0 cols] (row-major, stride d0) into OFF_BFULL.
__device__ __forceinline__ void reload_panel(char* smem, const __bf16* panel,
                                             unsigned d0, unsigned d1, int t, int wv) {
    __bf16* Bfull = (__bf16*)(smem + OFF_BFULL);
#ifdef HAVE_TDM
    if (wv == 0) {
        tdm_load_2d((unsigned int)(size_t)Bfull, panel, d0, d1, d0, d1, d0, 1u);
        __builtin_amdgcn_s_wait_tensorcnt(0);
    }
#else
    for (int i = t * 16; i < (int)(d0 * d1); i += 512 * 16) {
        #pragma unroll
        for (int j = 0; j < 16; ++j) Bfull[i + j] = panel[i + j];
    }
#endif
    __syncthreads();
}

// ---------------- WMMA kernels ----------------

// score = xn @ mn^T for 16 tokens x 256 k; argmax over k -> ind
// block: 512 threads = 16 waves; grid: HW/16; barrier-free inner K-loop
__global__ void k_score_argmax(const __bf16* __restrict__ xnT, const __bf16* __restrict__ mn,
                               int* __restrict__ ind, int b) {
    extern __shared__ char smem[];
    __bf16* Bfull = (__bf16*)(smem + OFF_BFULL);
    __bf16* As    = (__bf16*)(smem + OFF_AS);
    float*  Sc    = (float*)(smem + OFF_SC);
    const int t = threadIdx.x, lane = t & 31, wv = t >> 5;
    const int n0 = blockIdx.x * 16;
    const __bf16* xb = xnT + (size_t)b * CS * HW;

    load_panels(smem, mn, xb + n0, t, wv);

    v8f acc = {};
    #pragma unroll 4
    for (int kk = 0; kk < CS; kk += 32) {
        v16bf a, bb;
        #pragma unroll
        for (int e = 0; e < 16; ++e) a[e]  = As[(lane & 15) * CS + kk + a_kloc(e, lane)];
        #pragma unroll
        for (int e = 0; e < 16; ++e)
            bb[e] = Bfull[(wv * 16 + (lane & 15)) * CS + kk + b_kloc(e, lane)];
        acc = __builtin_amdgcn_wmma_f32_16x16x32_bf16(false, a, false, bb, (short)0, acc, false, false);
    }
    __syncthreads();
    #pragma unroll
    for (int v = 0; v < 8; ++v) {
        int M = v + ((lane >> 4) << 3);
        Sc[M * 256 + wv * 16 + (lane & 15)] = acc[v];
    }
    __syncthreads();
    if (t < 16) {   // first-max argmax, matches jnp.argmax
        float best = -1e30f; int bi = 0;
        for (int k = 0; k < NK; ++k) {
            float s = Sc[t * 256 + k];
            if (s > best) { best = s; bi = k; }
        }
        ind[n0 + t] = bi;
    }
}

__global__ void k_zero(float* __restrict__ esum, float* __restrict__ cnt) {
    int i = blockIdx.x * 256 + threadIdx.x;
    if (i < NK * CS) esum[i] = 0.f;
    if (i < NK)      cnt[i]  = 0.f;
}

// scatter-add raw token values into per-cluster sums (+ counts)
__global__ void k_scatter(const float* __restrict__ x, const int* __restrict__ ind,
                          float* __restrict__ esum, float* __restrict__ cnt, int b) {
    int gid = blockIdx.x * 256 + threadIdx.x;   // 0 .. HW*32-1
    int n  = gid >> 5;
    int ch = (gid & 31) * 16;
    int k  = ind[n];
    const float* xb = x + (size_t)b * CS * HW;
    float* er = esum + (size_t)k * CS;
    #pragma unroll
    for (int i = 0; i < 16; ++i)
        atomicAdd(&er[ch + i], xb[(size_t)(ch + i) * HW + n]);
    if ((gid & 31) == 0) atomicAdd(&cnt[k], 1.0f);
}

// EMA update; writes m_final[label], mem[b] (f32) and mem_bfT[b] (bf16, [CS][NK])
__global__ void k_update(float* __restrict__ mwork, const int* __restrict__ label,
                         const float* __restrict__ esum, const float* __restrict__ cnt,
                         float* __restrict__ mem, __bf16* __restrict__ mem_bfT, int b) {
    int i = blockIdx.x * 256 + threadIdx.x;     // over NK*CS
    int k = i / CS, c = i % CS;
    int lab = label[b];
    float* mrow = mwork + (size_t)lab * NK * CS;
    float mean = esum[i] / (cnt[k] + 1e-8f);
    float nv = mrow[i] * MAR + mean * (1.0f - MAR);
    mrow[i] = nv;
    mem[(size_t)b * NK * CS + i] = nv;
    mem_bfT[(size_t)b * CS * NK + (size_t)c * NK + k] = f2bf(nv);   // transposed copy
}

// normalize mem rows -> mnA bf16 ; grid = NB*NK
__global__ void k_norm_mem(const float* __restrict__ mem, __bf16* __restrict__ mnA) {
    const float* row = mem + (size_t)blockIdx.x * CS;
    float v0 = row[threadIdx.x], v1 = row[threadIdx.x + 256];
    __shared__ float red[256];
    red[threadIdx.x] = v0 * v0 + v1 * v1;
    __syncthreads();
    for (int s = 128; s > 0; s >>= 1) {
        if (threadIdx.x < s) red[threadIdx.x] += red[threadIdx.x + s];
        __syncthreads();
    }
    float inv = 1.0f / fmaxf(sqrtf(red[0]), 1e-12f);
    __bf16* o = mnA + (size_t)blockIdx.x * CS;
    o[threadIdx.x]       = f2bf(v0 * inv);
    o[threadIdx.x + 256] = f2bf(v1 * inv);
}

// attention: score (WMMA) -> softmax -> out = soft @ mem (WMMA); TDM-resident panels
__global__ void k_attn(const __bf16* __restrict__ xnT, const __bf16* __restrict__ mnA,
                       const __bf16* __restrict__ mem_bfT, float* __restrict__ outg, int b) {
    extern __shared__ char smem[];
    __bf16* Bfull = (__bf16*)(smem + OFF_BFULL);
    __bf16* As    = (__bf16*)(smem + OFF_AS);
    float*  Sc    = (float*)(smem + OFF_SC);
    __bf16* Sf    = (__bf16*)(smem + OFF_SF);
    const int t = threadIdx.x, lane = t & 31, wv = t >> 5;
    const int n0 = blockIdx.x * 16;
    const __bf16* xb = xnT + (size_t)b * CS * HW;

    // ---- P1: score tile [16 x 256], barrier-free K-loop ----
    load_panels(smem, mnA + (size_t)b * NK * CS, xb + n0, t, wv);
    v8f acc = {};
    #pragma unroll 4
    for (int kk = 0; kk < CS; kk += 32) {
        v16bf a, bb;
        #pragma unroll
        for (int e = 0; e < 16; ++e) a[e]  = As[(lane & 15) * CS + kk + a_kloc(e, lane)];
        #pragma unroll
        for (int e = 0; e < 16; ++e)
            bb[e] = Bfull[(wv * 16 + (lane & 15)) * CS + kk + b_kloc(e, lane)];
        acc = __builtin_amdgcn_wmma_f32_16x16x32_bf16(false, a, false, bb, (short)0, acc, false, false);
    }
    __syncthreads();
    #pragma unroll
    for (int v = 0; v < 8; ++v) {
        int M = v + ((lane >> 4) << 3);
        Sc[M * 256 + wv * 16 + (lane & 15)] = acc[v];
    }
    __syncthreads();

    // ---- P2: row softmax over k (fp32), emit bf16 ----
    if (t < 16) {
        float mx = -1e30f;
        for (int k = 0; k < NK; ++k) mx = fmaxf(mx, Sc[t * 256 + k]);
        float sum = 0.f;
        for (int k = 0; k < NK; ++k) { float e = __expf(Sc[t * 256 + k] - mx); Sc[t * 256 + k] = e; sum += e; }
        float inv = 1.0f / sum;
        for (int k = 0; k < NK; ++k) Sf[t * 256 + k] = f2bf(Sc[t * 256 + k] * inv);
    }
    __syncthreads();   // all P1 reads of Bfull complete before reload below

    // ---- P3: out[16 x 512] = soft[16 x 256] @ mem[256 x 512] ----
    // Bfull := mem_bfT[b] laid out [CS][NK] so B fragments are contiguous in k
    reload_panel(smem, mem_bfT + (size_t)b * CS * NK, NK, CS, t, wv);
    v8f o0 = {}, o1 = {};
    #pragma unroll 2
    for (int kk = 0; kk < NK; kk += 32) {
        v16bf a, b1, b2;
        #pragma unroll
        for (int e = 0; e < 16; ++e) a[e]  = Sf[(lane & 15) * 256 + kk + a_kloc(e, lane)];
        #pragma unroll
        for (int e = 0; e < 16; ++e)
            b1[e] = Bfull[(wv * 16 + (lane & 15)) * NK + kk + b_kloc(e, lane)];
        o0 = __builtin_amdgcn_wmma_f32_16x16x32_bf16(false, a, false, b1, (short)0, o0, false, false);
        #pragma unroll
        for (int e = 0; e < 16; ++e)
            b2[e] = Bfull[((wv + 16) * 16 + (lane & 15)) * NK + kk + b_kloc(e, lane)];
        o1 = __builtin_amdgcn_wmma_f32_16x16x32_bf16(false, a, false, b2, (short)0, o1, false, false);
    }
    float* ob = outg + (size_t)b * CS * HW;   // channel-major [CS][HW]
    #pragma unroll
    for (int v = 0; v < 8; ++v) {
        int M = v + ((lane >> 4) << 3);
        int N = lane & 15;
        ob[(size_t)(wv * 16 + N) * HW + n0 + M]        = o0[v];
        ob[(size_t)((wv + 16) * 16 + N) * HW + n0 + M] = o1[v];
    }
}

// ---------------- launcher ----------------
extern "C" void kernel_launch(void* const* d_in, const int* in_sizes, int n_in,
                              void* d_out, int out_size, void* d_ws, size_t ws_size,
                              hipStream_t stream) {
    const float* x     = (const float*)d_in[0];   // [B,CS,HW]
    const float* units = (const float*)d_in[1];   // [A,K,CS]
    const int*   label = (const int*)d_in[2];     // [B]

    float* m_final = (float*)d_out;                       // A*K*CS
    float* outg    = m_final + (size_t)NA * NK * CS;      // B*CS*HW

    char* ws = (char*)d_ws;
    size_t off = 0;
    __bf16* xnT    = (__bf16*)(ws + off); off += (size_t)NB * CS * HW * 2;  // 32 MB
    __bf16* mn     = (__bf16*)(ws + off); off += (size_t)NK * CS * 2;       // 256 KB
    int*    ind    = (int*)  (ws + off);  off += (size_t)HW * 4;            // 16 KB
    float*  cnt    = (float*)(ws + off);  off += (size_t)NK * 4;            // 1 KB
    float*  esum   = (float*)(ws + off);  off += (size_t)NK * CS * 4;       // 512 KB
    float*  mem    = (float*)(ws + off);  off += (size_t)NB * NK * CS * 4;  // 4 MB
    __bf16* mnA    = (__bf16*)(ws + off); off += (size_t)NB * NK * CS * 2;  // 2 MB
    __bf16* mem_bfT= (__bf16*)(ws + off); off += (size_t)NB * CS * NK * 2;  // 2 MB

    k_copy<<<(NA * NK * CS + 255) / 256, 256, 0, stream>>>(units, m_final, NA * NK * CS);
    k_norm_x<<<NB * (HW / 256), 256, 0, stream>>>(x, xnT);

    // sequential EMA recurrence over batch
    for (int b = 0; b < NB; ++b) {
        k_norm_m<<<NK, 256, 0, stream>>>(m_final, label, b, mn);
        k_score_argmax<<<HW / 16, 512, LDS_SCORE, stream>>>(xnT, mn, ind, b);
        k_zero<<<(NK * CS) / 256, 256, 0, stream>>>(esum, cnt);
        k_scatter<<<(HW * (CS / 16)) / 256, 256, 0, stream>>>(x, ind, esum, cnt, b);
        k_update<<<(NK * CS) / 256, 256, 0, stream>>>(m_final, label, esum, cnt, mem, mem_bfT, b);
    }

    // attention read
    k_norm_mem<<<NB * NK, 256, 0, stream>>>(mem, mnA);
    for (int b = 0; b < NB; ++b)
        k_attn<<<HW / 16, 512, LDS_ATTN, stream>>>(xnT, mnA, mem_bfT, outg, b);
}
